// PointerLSTM_7430293422955
// MI455X (gfx1250) — matrix operand order repaired and checked by reading, not compiled, not run on hardware
//
#include <hip/hip_runtime.h>
#include <math.h>

typedef __attribute__((ext_vector_type(2)))  float    v2f;
typedef __attribute__((ext_vector_type(8)))  float    v8f;
typedef __attribute__((ext_vector_type(8)))  _Float16 v8h;
typedef __attribute__((ext_vector_type(16))) _Float16 v16h;

#define NB 32
#define NS 256
#define NF 128
#define NH 256
#define NZ 1024  // 4*NH

__device__ __forceinline__ float sigf(float x) { return 1.0f / (1.0f + __expf(-x)); }

__device__ __forceinline__ v8f wmma_f32_k4(v2f a, v2f b, v8f c) {
    return __builtin_amdgcn_wmma_f32_16x16x4_f32(false, a, false, b, (short)0, c,
                                                 false, false);
}
__device__ __forceinline__ v8f wmma_f16_k32(v16h a, v16h b, v8f c) {
    return __builtin_amdgcn_wmma_f32_16x16x32_f16(false, a, false, b, (short)0, c,
                                                  false, false);
}
__device__ __forceinline__ v16h cat8(v8h lo, v8h hi) {
    return __builtin_shufflevector(lo, hi, 0, 1, 2, 3, 4, 5, 6, 7,
                                           8, 9, 10, 11, 12, 13, 14, 15);
}

// ---------------------------------------------------------------------------
// xz[b][j] = dot(x[b, S-1, :], Wk[:, j]) + bias[j]        (one-time, tiny)
// ---------------------------------------------------------------------------
__global__ void xz_kernel(const float* __restrict__ x, const float* __restrict__ Wk,
                          const float* __restrict__ bias, float* __restrict__ xz) {
    int idx = blockIdx.x * blockDim.x + threadIdx.x;
    int b = idx >> 10;
    int j = idx & 1023;
    const float* xrow = x + ((b * NS) + (NS - 1)) * NF;
    float acc = bias[j];
#pragma unroll 4
    for (int f = 0; f < NF; ++f) acc += xrow[f] * Wk[f * NZ + j];
    xz[b * NZ + j] = acc;
}

// ---------------------------------------------------------------------------
// One-time repack: Uk (f32, [k][col]) -> Ukp (f16, fragment-friendly):
// Ukp[((k>>5)*NZ + col)*32 + (k&31)]. A lane's 16-element B fragment for
// K-block kb, column col, half h is then 32 contiguous bytes.
// ---------------------------------------------------------------------------
__global__ void ukp_kernel(const float* __restrict__ Uk, _Float16* __restrict__ Ukp) {
    int i = blockIdx.x * blockDim.x + threadIdx.x;   // 256*1024 elements
    int k = i >> 10;
    int col = i & 1023;
    Ukp[(((size_t)(k >> 5) * NZ) + col) * 32 + (k & 31)] = (_Float16)Uk[i];
}

// ---------------------------------------------------------------------------
// w2eT[b][h][s] = sum_f x[b][s][f] * W2[f][h]  (f32 WMMA, one-time, stored
// transposed so attention reads coalesced along s).
// ---------------------------------------------------------------------------
__global__ void w2e_kernel(const float* __restrict__ x, const float* __restrict__ W2,
                           float* __restrict__ w2eT) {
    int wid  = (blockIdx.x * blockDim.x + threadIdx.x) >> 5;  // 8192 waves
    int lane = threadIdx.x & 31;
    int b    = wid >> 8;
    int rem  = wid & 255;
    int stile = (rem >> 4) << 4;
    int htile = (rem & 15) << 4;
    int half  = lane >> 4;
    int ln    = lane & 15;
    int kb    = half * 2;

    v8f acc = {};
    const float* xa = x + ((b * NS) + (stile + ln)) * NF;
    for (int k = 0; k < NF; k += 4) {
        v2f a;  a.x  = xa[k + kb];
                a.y  = xa[k + kb + 1];
        v2f bb; bb.x = W2[(k + kb)     * NH + htile + ln];
                bb.y = W2[(k + kb + 1) * NH + htile + ln];
        acc = wmma_f32_k4(a, bb, acc);
    }
    int srow = stile + half * 8;
    int h    = htile + ln;
#pragma unroll
    for (int r = 0; r < 8; ++r)
        w2eT[((b * NH) + h) * NS + (srow + r)] = acc[r];
}

// ---------------------------------------------------------------------------
// Persistent LSTM recurrence: ONE 512-thread workgroup (16 waves) runs all
// 256 steps using V_WMMA_F32_16X16X32_F16 (8x FLOP/instr vs f32 WMMA).
// Each wave owns one 16-wide hh tile and computes BOTH batch tiles for all
// 4 gates (8 accumulators) so every Uk fragment is loaded once, used twice.
// h ping-pongs in LDS as f16 (A fragments = 2x ds_load_b128 per matrix);
// c stays in f32 registers; xz staged in LDS.
// ---------------------------------------------------------------------------
__global__ void __launch_bounds__(512) recurrence_kernel(
    const float* __restrict__ xz, const _Float16* __restrict__ Ukp,
    float* __restrict__ hall) {
    __shared__ _Float16 hls[2][NB * NH];   // 2 x 16 KB ping-pong h (f16)
    __shared__ float    xzs[NB * NZ];      // 128 KB

    int tid  = threadIdx.x;
    int wid  = tid >> 5;   // 0..15 -> Ntile
    int lane = tid & 31;

    for (int i = tid; i < NB * NZ; i += 512) xzs[i] = xz[i];
    for (int i = tid; i < NB * NH; i += 512) hls[0][i] = (_Float16)0.0f;
    __syncthreads();

    int Ntile = wid << 4;
    int half  = lane >> 4;
    int ln    = lane & 15;
    int h8    = half << 3;   // unused; A fragments use h16
    int h16   = half << 4;   // B-fragment K sub-offset
    int col   = Ntile + ln;  // hh column

    float creg0[8], creg1[8];
#pragma unroll
    for (int r = 0; r < 8; ++r) { creg0[r] = 0.0f; creg1[r] = 0.0f; }

    int cur = 0;
    for (int t = 0; t < NS; ++t) {
        const _Float16* h0row = &hls[cur][ln * NH];          // M = 0..15
        const _Float16* h1row = &hls[cur][(16 + ln) * NH];   // M = 16..31
        v8f acc00 = {}, acc01 = {}, acc10 = {}, acc11 = {};
        v8f acc20 = {}, acc21 = {}, acc30 = {}, acc31 = {};
        for (int k = 0; k < NH; k += 32) {
            // A fragments: elements 0..15 <-> K half*16+e
            // -> two aligned 16B contiguous LDS reads per matrix.
            v16h a0 = cat8(*(const v8h*)(h0row + k + h16),
                           *(const v8h*)(h0row + k + h16 + 8));
            v16h a1 = cat8(*(const v8h*)(h1row + k + h16),
                           *(const v8h*)(h1row + k + h16 + 8));
            const _Float16* ub = Ukp + ((size_t)(k >> 5) * NZ) * 32 + h16;
            v16h b0 = *(const v16h*)(ub + (0 * NH + col) * 32);
            v16h b1 = *(const v16h*)(ub + (1 * NH + col) * 32);
            v16h b2 = *(const v16h*)(ub + (2 * NH + col) * 32);
            v16h b3 = *(const v16h*)(ub + (3 * NH + col) * 32);
            acc00 = wmma_f16_k32(a0, b0, acc00);
            acc01 = wmma_f16_k32(a1, b0, acc01);
            acc10 = wmma_f16_k32(a0, b1, acc10);
            acc11 = wmma_f16_k32(a1, b1, acc11);
            acc20 = wmma_f16_k32(a0, b2, acc20);
            acc21 = wmma_f16_k32(a1, b2, acc21);
            acc30 = wmma_f16_k32(a0, b3, acc30);
            acc31 = wmma_f16_k32(a1, b3, acc31);
        }
        int nxt = cur ^ 1;
#pragma unroll
        for (int r = 0; r < 8; ++r) {        // batch tile 0 (b = 0..15)
            int bi = half * 8 + r;
            const float* xzr = xzs + bi * NZ;
            float zi = acc00[r] + xzr[col];
            float zf = acc10[r] + xzr[col + NH];
            float zg = acc20[r] + xzr[col + 2 * NH];
            float zo = acc30[r] + xzr[col + 3 * NH];
            float cn = sigf(zf) * creg0[r] + sigf(zi) * tanhf(zg);
            float hn = sigf(zo) * tanhf(cn);
            creg0[r] = cn;
            hls[nxt][bi * NH + col] = (_Float16)hn;
            hall[((t * NB) + bi) * NH + col] = hn;
        }
#pragma unroll
        for (int r = 0; r < 8; ++r) {        // batch tile 1 (b = 16..31)
            int bi = 16 + half * 8 + r;
            const float* xzr = xzs + bi * NZ;
            float zi = acc01[r] + xzr[col];
            float zf = acc11[r] + xzr[col + NH];
            float zg = acc21[r] + xzr[col + 2 * NH];
            float zo = acc31[r] + xzr[col + 3 * NH];
            float cn = sigf(zf) * creg1[r] + sigf(zi) * tanhf(zg);
            float hn = sigf(zo) * tanhf(cn);
            creg1[r] = cn;
            hls[nxt][bi * NH + col] = (_Float16)hn;
            hall[((t * NB) + bi) * NH + col] = hn;
        }
        __syncthreads();   // hls[nxt] complete for all waves
        cur = nxt;
    }
}

// ---------------------------------------------------------------------------
// Q = hall @ W1 : (8192 x 256) @ (256 x 256), f32 WMMA, one-time.
// ---------------------------------------------------------------------------
__global__ void q_kernel(const float* __restrict__ hall, const float* __restrict__ W1,
                         float* __restrict__ Q) {
    int wid  = (blockIdx.x * blockDim.x + threadIdx.x) >> 5;  // 8192 waves
    int lane = threadIdx.x & 31;
    int rtile = (wid >> 4) << 4;
    int htile = (wid & 15) << 4;
    int half  = lane >> 4;
    int ln    = lane & 15;
    int kb    = half * 2;

    v8f acc = {};
    const float* xa = hall + (rtile + ln) * NH;
    for (int k = 0; k < NH; k += 4) {
        v2f a;  a.x  = xa[k + kb];
                a.y  = xa[k + kb + 1];
        v2f bb; bb.x = W1[(k + kb)     * NH + htile + ln];
                bb.y = W1[(k + kb + 1) * NH + htile + ln];
        acc = wmma_f32_k4(a, bb, acc);
    }
    int h = htile + ln;
#pragma unroll
    for (int r = 0; r < 8; ++r)
        Q[(rtile + half * 8 + r) * NH + h] = acc[r];
}

// ---------------------------------------------------------------------------
// Attention + softmax for ALL steps at once: grid = 8192 WGs, one per (t,b).
// ---------------------------------------------------------------------------
__global__ void attn_kernel(const float* __restrict__ Q, const float* __restrict__ w2eT,
                            const float* __restrict__ V, float* __restrict__ out) {
    __shared__ float qv[NH];
    __shared__ float vv[NH];
    __shared__ float red[NS];
    int tb  = blockIdx.x;        // t*NB + b
    int t   = tb >> 5;
    int b   = tb & 31;
    int tid = threadIdx.x;

    qv[tid] = Q[tb * NH + tid];
    vv[tid] = V[tid];
    __syncthreads();

    float sc = 0.f;
    const float* wb = w2eT + (size_t)(b * NH) * NS + tid;
#pragma unroll 8
    for (int h = 0; h < NH; ++h) {
        if ((h & 7) == 0) __builtin_prefetch(wb + (h + 8) * NS, 0, 1);
        sc += tanhf(qv[h] + wb[h * NS]) * vv[h];
    }

    red[tid] = sc;
    __syncthreads();
    for (int off = 128; off > 0; off >>= 1) {
        if (tid < off) red[tid] = fmaxf(red[tid], red[tid + off]);
        __syncthreads();
    }
    float m = red[0];
    __syncthreads();
    float e = __expf(sc - m);
    red[tid] = e;
    __syncthreads();
    for (int off = 128; off > 0; off >>= 1) {
        if (tid < off) red[tid] += red[tid + off];
        __syncthreads();
    }
    float s = red[0];
    out[((b * NS) + t) * NS + tid] = e / s;
}

// ---------------------------------------------------------------------------
extern "C" void kernel_launch(void* const* d_in, const int* in_sizes, int n_in,
                              void* d_out, int out_size, void* d_ws, size_t ws_size,
                              hipStream_t stream) {
    const float* x    = (const float*)d_in[0];
    const float* Wk   = (const float*)d_in[1];
    const float* Uk   = (const float*)d_in[2];
    const float* bias = (const float*)d_in[3];
    const float* W1   = (const float*)d_in[4];
    const float* W2   = (const float*)d_in[5];
    const float* V    = (const float*)d_in[6];
    float* out = (float*)d_out;

    float*    w2eT = (float*)d_ws;                 // 32*256*256
    float*    xz   = w2eT + NB * NH * NS;          // 32*1024
    float*    hall = xz + NB * NZ;                 // 256*32*256
    float*    Q    = hall + NS * NB * NH;          // 256*32*256
    _Float16* Ukp  = (_Float16*)(Q + NS * NB * NH);// 256*1024 f16 (repacked)

    xz_kernel<<<(NB * NZ) / 256, 256, 0, stream>>>(x, Wk, bias, xz);
    ukp_kernel<<<(NH * NZ) / 256, 256, 0, stream>>>(Uk, Ukp);
    w2e_kernel<<<(NB * (NS / 16) * (NH / 16) * 32) / 256, 256, 0, stream>>>(x, W2, w2eT);
    recurrence_kernel<<<1, 512, 0, stream>>>(xz, Ukp, hall);
    q_kernel<<<((NS * NB / 16) * (NH / 16) * 32) / 256, 256, 0, stream>>>(hall, W1, Q);
    attn_kernel<<<NS * NB, 256, 0, stream>>>(Q, w2eT, V, out);
}
